// ProgramEncoder_42288247997028
// MI455X (gfx1250) — compile-verified, matrix operand order checked by reading, not run on hardware
//
#include <hip/hip_runtime.h>
#include <hip/hip_bf16.h>
#include <math.h>

// ---------------- problem constants ----------------
#define BSZ   16
#define SEQL  256
#define NIMG  256
#define DMODEL 512
#define NHEAD 8
#define DHEAD 256
#define HD    2048   // NHEAD*DHEAD
#define NLAY  6
#define EMB   300
#define TOKS  8
#define KIN   2400   // TOKS*EMB
#define TKNS  4096   // BSZ*SEQL

// ---------------- WMMA types ----------------
typedef __attribute__((ext_vector_type(16))) _Float16 v16h;
typedef __attribute__((ext_vector_type(8)))  float    v8f;

// GEMM tiling: 128x128x32 block, 256 threads = 8 wave32, wave tile 32x64 (2x4 wmma)
#define BM 128
#define BN 128
#define BK 32
#define SSTR 40   // LDS row stride in halves (32 + 8 pad), 80B => 16B-aligned rows

__device__ __forceinline__ _Float16 f2h(float f) { return (_Float16)f; }

// CDNA5 async global->LDS copy (ASYNCcnt-tracked, no VGPR staging).
// lds = workgroup-relative LDS byte address (flat-pointer low 32 bits, ISA 10.2).
__device__ __forceinline__ void async_copy_b128(unsigned lds, const _Float16* g) {
    asm volatile("global_load_async_to_lds_b128 %0, %1, off"
                 :: "v"(lds), "v"(g) : "memory");
}
__device__ __forceinline__ void wait_async0() {
    asm volatile("s_wait_asynccnt 0x0" ::: "memory");
}
__device__ __forceinline__ unsigned lds_addr(const void* p) {
    return (unsigned)(unsigned long long)(uintptr_t)p;
}

// Load one 16x16(xK=32) f16 WMMA fragment from an LDS tile stored row-major
// (rows = M for A, rows = N for B) with row stride SSTR halves.
// ISA layout: lanes 0-15 row=lane, elems {K0..7, K16..23}; lanes 16-31 row=lane-16,
// elems {K8..15, K24..31}.
__device__ __forceinline__ v16h load_frag(const _Float16* rowbase, int lane) {
    union { v16h v; _Float16 h[16]; } u;
    const _Float16* p = rowbase + (lane & 15) * SSTR + ((lane >> 4) << 3);
    __builtin_memcpy(&u.h[0], p, 16);        // 8 halves: K 0..7 (or 8..15)
    __builtin_memcpy(&u.h[8], p + 16, 16);   // 8 halves: K 16..23 (or 24..31)
    return u.v;
}

// ---------------------------------------------------------------------------
// Unified batched GEMM: C = act(alpha * A*B + bias + residual)
//   A: (M,K) f16 row-major, lda
//   B: (N,K) f16 row-major, ldb  (pre-transposed weight / K-matrix / V^T)
//   batch z: off = (z/inner)*s0 + (z%inner)*s1   (element offsets)
//   outputs: optional fp32 Cf and/or fp16 Ch at ldc
//   act: 0 = none, 1 = exact GELU
// Staging: double-buffered LDS fed by global_load_async_to_lds_b128; the next
// K-tile's async copies overlap the current tile's 8 WMMAs.
// ---------------------------------------------------------------------------
__global__ __launch_bounds__(256)
void gemm_f16_kernel(const _Float16* __restrict__ A, long long lda, long long sA0, long long sA1,
                     const _Float16* __restrict__ B, long long ldb, long long sB0, long long sB1,
                     float* __restrict__ Cf, _Float16* __restrict__ Ch,
                     long long ldc, long long sC0, long long sC1,
                     const float* __restrict__ bias, const float* __restrict__ resid,
                     int M, int N, int K, int inner, float alpha, int act) {
    __shared__ _Float16 sA[2][BM * SSTR];
    __shared__ _Float16 sB[2][BN * SSTR];

    const int tid  = threadIdx.x;
    const int lane = tid & 31;
    const int wid  = tid >> 5;   // 0..7
    const int wm   = wid >> 1;   // 0..3 -> 32-row slice
    const int wn   = wid & 1;    // 0..1 -> 64-col slice

    const int z  = blockIdx.z;
    const long long offA = (long long)(z / inner) * sA0 + (long long)(z % inner) * sA1;
    const long long offB = (long long)(z / inner) * sB0 + (long long)(z % inner) * sB1;
    const long long offC = (long long)(z / inner) * sC0 + (long long)(z % inner) * sC1;

    const int m0 = blockIdx.y * BM;
    const int n0 = blockIdx.x * BN;

    v8f acc[2][4];
    const v8f vzero = {0.f, 0.f, 0.f, 0.f, 0.f, 0.f, 0.f, 0.f};
#pragma unroll
    for (int i = 0; i < 2; ++i)
#pragma unroll
        for (int j = 0; j < 4; ++j) acc[i][j] = vzero;

    // each thread owns one 16-half (32B) segment of one row of each tile
    const int rA = tid >> 1;            // 0..127
    const int cA = (tid & 1) << 4;      // 0 or 16 halves
    const _Float16* gA = A + offA + (long long)(m0 + rA) * lda + cA;
    const _Float16* gB = B + offB + (long long)(n0 + rA) * ldb + cA;
    const unsigned la[2] = { lds_addr(&sA[0][rA * SSTR + cA]), lds_addr(&sA[1][rA * SSTR + cA]) };
    const unsigned lb[2] = { lds_addr(&sB[0][rA * SSTR + cA]), lds_addr(&sB[1][rA * SSTR + cA]) };

    auto issue_tile = [&](int buf, int k0) {
        async_copy_b128(la[buf],      gA + k0);
        async_copy_b128(la[buf] + 16, gA + k0 + 8);
        async_copy_b128(lb[buf],      gB + k0);
        async_copy_b128(lb[buf] + 16, gB + k0 + 8);
    };

    // prologue: fill buffer 0
    issue_tile(0, 0);
    wait_async0();
    __syncthreads();

    int buf = 0;
    for (int k0 = 0; k0 < K; k0 += BK) {
        // overlap: start async copies of the next K-tile into the other buffer
        if (k0 + BK < K) issue_tile(buf ^ 1, k0 + BK);
        if (k0 + 2 * BK < K) {
            __builtin_prefetch((const void*)(gA + k0 + 2 * BK), 0, 1);
            __builtin_prefetch((const void*)(gB + k0 + 2 * BK), 0, 1);
        }

        // ---- 8 WMMAs on the current buffer ----
        v16h af[2], bfr[4];
#pragma unroll
        for (int i = 0; i < 2; ++i)
            af[i] = load_frag(&sA[buf][(wm * 32 + i * 16) * SSTR], lane);
#pragma unroll
        for (int j = 0; j < 4; ++j)
            bfr[j] = load_frag(&sB[buf][(wn * 64 + j * 16) * SSTR], lane);
#pragma unroll
        for (int i = 0; i < 2; ++i)
#pragma unroll
            for (int j = 0; j < 4; ++j)
                acc[i][j] = __builtin_amdgcn_wmma_f32_16x16x32_f16(
                    false, af[i], false, bfr[j], (short)0, acc[i][j], false, false);

        // own async copies done, then all waves' copies visible after barrier
        wait_async0();
        __syncthreads();
        buf ^= 1;
    }

    // ---- epilogue: C/D layout => n = lane&15, m = e + 8*(lane>>4) ----
#pragma unroll
    for (int i = 0; i < 2; ++i) {
#pragma unroll
        for (int j = 0; j < 4; ++j) {
            const int rb = m0 + wm * 32 + i * 16 + ((lane >> 4) << 3);
            const int c  = n0 + wn * 64 + j * 16 + (lane & 15);
            const float bval = bias ? bias[c] : 0.0f;
#pragma unroll
            for (int e = 0; e < 8; ++e) {
                const int r = rb + e;
                float v = acc[i][j][e] * alpha + bval;
                const long long idx = offC + (long long)r * ldc + c;
                if (resid) v += resid[idx];
                if (act == 1) v = 0.5f * v * (1.0f + erff(v * 0.70710678118654752f));
                if (Cf) Cf[idx] = v;
                if (Ch) Ch[idx] = f2h(v);
            }
        }
    }
}

// ---------------- embedding gather: A0[t, tok*300+d] = embed[program[t,tok], d] ----
__global__ __launch_bounds__(256)
void gather_embed_kernel(const int* __restrict__ program, const float* __restrict__ embed,
                         _Float16* __restrict__ A0) {
    const int t = blockIdx.x;  // 0..4095
    for (int j = threadIdx.x; j < KIN; j += blockDim.x) {
        const int tok = j / EMB;
        const int d   = j - tok * EMB;
        const int pid = program[t * TOKS + tok];
        A0[(long long)t * KIN + j] = f2h(embed[pid * EMB + d]);
    }
}

// ---------------- weight convert+transpose: Wt(N,K) f16 <- W(K,N) f32 ----
__global__ __launch_bounds__(256)
void convert_wT_kernel(const float* __restrict__ W, _Float16* __restrict__ Wt, int K, int N) {
    const int idx = blockIdx.x * 256 + threadIdx.x;
    if (idx < K * N) {
        const int k = idx % K;
        const int n = idx / K;
        Wt[(long long)n * K + k] = f2h(W[(long long)k * N + n]);
    }
}

// ---------------- V transpose: vT[(b*H+h)*DHEAD + n][k] = V[b*SEQL+k][h*DHEAD+n] ----
__global__ __launch_bounds__(256)
void transpose_v_kernel(const _Float16* __restrict__ V, _Float16* __restrict__ vT) {
    const long long i = (long long)blockIdx.x * 256 + threadIdx.x;  // over TKNS*HD
    const int k  = (int)(i & (SEQL - 1));
    const long long r = i >> 8;              // (b*H+h)*DHEAD + n
    const int n  = (int)(r & (DHEAD - 1));
    const int bh = (int)(r >> 8);
    const int b  = bh >> 3;                  // /NHEAD
    const int h  = bh & (NHEAD - 1);
    vT[i] = V[(long long)(b * SEQL + k) * HD + h * DHEAD + n];
}

// ---------------- fp32 -> fp16 elementwise ----
__global__ __launch_bounds__(256)
void convert_h_kernel(const float* __restrict__ x, _Float16* __restrict__ y, long long n) {
    const long long i = (long long)blockIdx.x * 256 + threadIdx.x;
    if (i < n) y[i] = f2h(x[i]);
}

// ---------------- LayerNorm (D=512) -> fp16, one row per block ----
__global__ __launch_bounds__(256)
void layernorm_kernel(const float* __restrict__ x, const float* __restrict__ g,
                      const float* __restrict__ b, _Float16* __restrict__ out) {
    const int row = blockIdx.x;
    const int tid = threadIdx.x;
    const float* xr = x + (long long)row * DMODEL;
    const float v0 = xr[tid];
    const float v1 = xr[tid + 256];
    __shared__ float red[256];
    red[tid] = v0 + v1;
    __syncthreads();
    for (int st = 128; st > 0; st >>= 1) {
        if (tid < st) red[tid] += red[tid + st];
        __syncthreads();
    }
    const float mu = red[0] * (1.0f / DMODEL);
    __syncthreads();
    const float d0 = v0 - mu, d1 = v1 - mu;
    red[tid] = d0 * d0 + d1 * d1;
    __syncthreads();
    for (int st = 128; st > 0; st >>= 1) {
        if (tid < st) red[tid] += red[tid + st];
        __syncthreads();
    }
    const float rs = rsqrtf(red[0] * (1.0f / DMODEL) + 1e-5f);
    out[(long long)row * DMODEL + tid]       = f2h(d0 * rs * g[tid] + b[tid]);
    out[(long long)row * DMODEL + tid + 256] = f2h(d1 * rs * g[tid + 256] + b[tid + 256]);
}

// ---------------- masked softmax with +1 denominator (cols = 256) ----
// block = one score row; bid = (b*NHEAD + h)*SEQL + q
__global__ __launch_bounds__(256)
void softmax_kernel(const float* __restrict__ S, _Float16* __restrict__ P,
                    const int* __restrict__ adj, int useMask) {
    const int bid = blockIdx.x;
    const int tid = threadIdx.x;
    float v = S[(long long)bid * 256 + tid];
    if (useMask) {
        const int q = bid % SEQL;
        const int b = bid / (NHEAD * SEQL);
        const int a = adj[((long long)b * SEQL + q) * SEQL + tid];
        v += (1.0f - (float)a) * -1e6f;
    }
    __shared__ float red[256];
    red[tid] = v;
    __syncthreads();
    for (int st = 128; st > 0; st >>= 1) {
        if (tid < st) red[tid] = fmaxf(red[tid], red[tid + st]);
        __syncthreads();
    }
    const float m = red[0];
    __syncthreads();
    const float e = __expf(v - m);
    red[tid] = e;
    __syncthreads();
    for (int st = 128; st > 0; st >>= 1) {
        if (tid < st) red[tid] += red[tid + st];
        __syncthreads();
    }
    P[(long long)bid * 256 + tid] = f2h(e / (1.0f + red[0]));
}

// ---------------- host-side launch helpers ----------------
static inline void launch_gemm(hipStream_t s,
                               const _Float16* A, long long lda, long long a0, long long a1,
                               const _Float16* B, long long ldb, long long b0, long long b1,
                               float* Cf, _Float16* Ch, long long ldc, long long c0, long long c1,
                               const float* bias, const float* resid,
                               int M, int N, int K, int inner, int batch, float alpha, int act) {
    dim3 grid(N / BN, M / BM, batch);
    gemm_f16_kernel<<<grid, 256, 0, s>>>(A, lda, a0, a1, B, ldb, b0, b1,
                                         Cf, Ch, ldc, c0, c1, bias, resid,
                                         M, N, K, inner, alpha, act);
}

static inline void launch_wT(hipStream_t s, const float* W, _Float16* Wt, int K, int N) {
    int total = K * N;
    convert_wT_kernel<<<(total + 255) / 256, 256, 0, s>>>(W, Wt, K, N);
}

extern "C" void kernel_launch(void* const* d_in, const int* in_sizes, int n_in,
                              void* d_out, int out_size, void* d_ws, size_t ws_size,
                              hipStream_t stream) {
    (void)in_sizes; (void)n_in; (void)out_size; (void)ws_size;

    // ---- inputs (setup_inputs order) ----
    const int*   program = (const int*)  d_in[0];
    const int*   pro_adj = (const int*)  d_in[1];
    const float* img     = (const float*)d_in[2];
    const float* embed   = (const float*)d_in[3];
    const float* W_in    = (const float*)d_in[4];
    const float* b_in    = (const float*)d_in[5];
    const float* ln1_g   = (const float*)d_in[6];
    const float* ln1_b   = (const float*)d_in[7];
    const float* ln2_g   = (const float*)d_in[8];
    const float* ln2_b   = (const float*)d_in[9];
    const float* Wq = (const float*)d_in[10]; const float* bq = (const float*)d_in[11];
    const float* Wk = (const float*)d_in[12]; const float* bk = (const float*)d_in[13];
    const float* Wv = (const float*)d_in[14]; const float* bv = (const float*)d_in[15];
    const float* Wo = (const float*)d_in[16]; const float* bo = (const float*)d_in[17];
    const float* fW1 = (const float*)d_in[18]; const float* fb1 = (const float*)d_in[19];
    const float* fW2 = (const float*)d_in[20]; const float* fb2 = (const float*)d_in[21];
    const float* cWq = (const float*)d_in[22]; const float* cbq = (const float*)d_in[23];
    const float* cWk = (const float*)d_in[24]; const float* cbk = (const float*)d_in[25];
    const float* cWv = (const float*)d_in[26]; const float* cbv = (const float*)d_in[27];
    const float* cWo = (const float*)d_in[28]; const float* cbo = (const float*)d_in[29];

    // ---- workspace carve ----
    char* ws = (char*)d_ws;
    auto carve = [&](size_t bytes) -> void* {
        void* p = (void*)ws;
        ws += (bytes + 255) & ~(size_t)255;
        return p;
    };
    _Float16* A0h  = (_Float16*)carve((size_t)TKNS * KIN * 2);
    float*    xf   = (float*)   carve((size_t)TKNS * DMODEL * 4);
    _Float16* hh   = (_Float16*)carve((size_t)TKNS * DMODEL * 2);
    _Float16* th   = (_Float16*)carve((size_t)TKNS * DMODEL * 2);
    _Float16* xh   = (_Float16*)carve((size_t)TKNS * DMODEL * 2);
    _Float16* imgh = (_Float16*)carve((size_t)BSZ * NIMG * DMODEL * 2);
    _Float16* qh   = (_Float16*)carve((size_t)TKNS * HD * 2);
    _Float16* kh   = (_Float16*)carve((size_t)TKNS * HD * 2);
    _Float16* vh   = (_Float16*)carve((size_t)TKNS * HD * 2);
    _Float16* vTh  = (_Float16*)carve((size_t)TKNS * HD * 2);
    _Float16* oh   = (_Float16*)carve((size_t)TKNS * HD * 2);
    float*    Sf   = (float*)   carve((size_t)BSZ * NHEAD * SEQL * SEQL * 4);
    _Float16* Ph   = (_Float16*)carve((size_t)BSZ * NHEAD * SEQL * SEQL * 2);
    _Float16* wT   = (_Float16*)carve((size_t)KIN * DMODEL * 2);  // largest weight

    const long long ZB = 0;
    const int NBH = BSZ * NHEAD;                  // 128 attention sub-problems
    const long long sQb = (long long)SEQL * HD;   // per-batch stride in q/k/v/o
    const long long sQh = DHEAD;                  // per-head stride
    const long long sSb = (long long)NHEAD * SEQL * SEQL;
    const long long sSh = (long long)SEQL * SEQL;
    const long long sVb = (long long)NHEAD * DHEAD * SEQL;  // vT per-batch
    const long long sVh = (long long)DHEAD * SEQL;          // vT per-head
    const float invsq = 1.0f / 16.0f;             // 1/sqrt(DHEAD)

    // ---- embedding: x = gather(embed, program) @ W_in + b_in ----
    gather_embed_kernel<<<TKNS, 256, 0, stream>>>(program, embed, A0h);
    launch_wT(stream, W_in, wT, KIN, DMODEL);
    launch_gemm(stream, A0h, KIN, ZB, ZB, wT, KIN, ZB, ZB,
                xf, nullptr, DMODEL, ZB, ZB, b_in, nullptr,
                TKNS, DMODEL, KIN, 1, 1, 1.0f, 0);

    // img features fp16 (constant across layers)
    {
        long long n = (long long)BSZ * NIMG * DMODEL;
        convert_h_kernel<<<(int)((n + 255) / 256), 256, 0, stream>>>(img, imgh, n);
    }

    for (int i = 0; i < NLAY; ++i) {
        // ===== self-attention =====
        layernorm_kernel<<<TKNS, 256, 0, stream>>>(xf, ln1_g + i * DMODEL, ln1_b + i * DMODEL, hh);

        launch_wT(stream, Wq + (size_t)i * DMODEL * HD, wT, DMODEL, HD);
        launch_gemm(stream, hh, DMODEL, ZB, ZB, wT, DMODEL, ZB, ZB,
                    nullptr, qh, HD, ZB, ZB, bq + (size_t)i * HD, nullptr,
                    TKNS, HD, DMODEL, 1, 1, 1.0f, 0);
        launch_wT(stream, Wk + (size_t)i * DMODEL * HD, wT, DMODEL, HD);
        launch_gemm(stream, hh, DMODEL, ZB, ZB, wT, DMODEL, ZB, ZB,
                    nullptr, kh, HD, ZB, ZB, bk + (size_t)i * HD, nullptr,
                    TKNS, HD, DMODEL, 1, 1, 1.0f, 0);
        launch_wT(stream, Wv + (size_t)i * DMODEL * HD, wT, DMODEL, HD);
        launch_gemm(stream, hh, DMODEL, ZB, ZB, wT, DMODEL, ZB, ZB,
                    nullptr, vh, HD, ZB, ZB, bv + (size_t)i * HD, nullptr,
                    TKNS, HD, DMODEL, 1, 1, 1.0f, 0);
        transpose_v_kernel<<<(TKNS * HD) / 256, 256, 0, stream>>>(vh, vTh);

        // S = Q K^T / sqrt(dh)  (K-matrix rows are exactly the N x K operand)
        launch_gemm(stream, qh, HD, sQb, sQh, kh, HD, sQb, sQh,
                    Sf, nullptr, SEQL, sSb, sSh, nullptr, nullptr,
                    SEQL, SEQL, DHEAD, NHEAD, NBH, invsq, 0);
        softmax_kernel<<<NBH * SEQL, 256, 0, stream>>>(Sf, Ph, pro_adj, 1);
        // O = P V   (B = vT, N x K layout)
        launch_gemm(stream, Ph, SEQL, sSb, sSh, vTh, SEQL, sVb, sVh,
                    nullptr, oh, HD, sQb, sQh, nullptr, nullptr,
                    SEQL, DHEAD, SEQL, NHEAD, NBH, 1.0f, 0);
        // x += O Wo + bo
        launch_wT(stream, Wo + (size_t)i * HD * DMODEL, wT, HD, DMODEL);
        launch_gemm(stream, oh, HD, ZB, ZB, wT, HD, ZB, ZB,
                    xf, nullptr, DMODEL, ZB, ZB, bo + (size_t)i * DMODEL, xf,
                    TKNS, DMODEL, HD, 1, 1, 1.0f, 0);

        // ===== FFN =====
        layernorm_kernel<<<TKNS, 256, 0, stream>>>(xf, ln2_g + i * DMODEL, ln2_b + i * DMODEL, hh);
        launch_wT(stream, fW1 + (size_t)i * DMODEL * DMODEL, wT, DMODEL, DMODEL);
        launch_gemm(stream, hh, DMODEL, ZB, ZB, wT, DMODEL, ZB, ZB,
                    nullptr, th, DMODEL, ZB, ZB, fb1 + (size_t)i * DMODEL, nullptr,
                    TKNS, DMODEL, DMODEL, 1, 1, 1.0f, 1 /*GELU*/);
        launch_wT(stream, fW2 + (size_t)i * DMODEL * DMODEL, wT, DMODEL, DMODEL);
        launch_gemm(stream, th, DMODEL, ZB, ZB, wT, DMODEL, ZB, ZB,
                    xf, nullptr, DMODEL, ZB, ZB, fb2 + (size_t)i * DMODEL, xf,
                    TKNS, DMODEL, DMODEL, 1, 1, 1.0f, 0);

        // ===== cross-attention (query = x, kv = img; no LN, no mask) =====
        {
            long long n = (long long)TKNS * DMODEL;
            convert_h_kernel<<<(int)((n + 255) / 256), 256, 0, stream>>>(xf, xh, n);
        }
        launch_wT(stream, cWq + (size_t)i * DMODEL * HD, wT, DMODEL, HD);
        launch_gemm(stream, xh, DMODEL, ZB, ZB, wT, DMODEL, ZB, ZB,
                    nullptr, qh, HD, ZB, ZB, cbq + (size_t)i * HD, nullptr,
                    TKNS, HD, DMODEL, 1, 1, 1.0f, 0);
        launch_wT(stream, cWk + (size_t)i * DMODEL * HD, wT, DMODEL, HD);
        launch_gemm(stream, imgh, DMODEL, ZB, ZB, wT, DMODEL, ZB, ZB,
                    nullptr, kh, HD, ZB, ZB, cbk + (size_t)i * HD, nullptr,
                    TKNS, HD, DMODEL, 1, 1, 1.0f, 0);
        launch_wT(stream, cWv + (size_t)i * DMODEL * HD, wT, DMODEL, HD);
        launch_gemm(stream, imgh, DMODEL, ZB, ZB, wT, DMODEL, ZB, ZB,
                    nullptr, vh, HD, ZB, ZB, cbv + (size_t)i * HD, nullptr,
                    TKNS, HD, DMODEL, 1, 1, 1.0f, 0);
        transpose_v_kernel<<<(TKNS * HD) / 256, 256, 0, stream>>>(vh, vTh);

        launch_gemm(stream, qh, HD, sQb, sQh, kh, HD, sQb, sQh,
                    Sf, nullptr, NIMG, sSb, sSh, nullptr, nullptr,
                    SEQL, NIMG, DHEAD, NHEAD, NBH, invsq, 0);
        softmax_kernel<<<NBH * SEQL, 256, 0, stream>>>(Sf, Ph, nullptr, 0);
        launch_gemm(stream, Ph, NIMG, sSb, sSh, vTh, NIMG, sVb, sVh,
                    nullptr, oh, HD, sQb, sQh, nullptr, nullptr,
                    SEQL, DHEAD, NIMG, NHEAD, NBH, 1.0f, 0);
        launch_wT(stream, cWo + (size_t)i * HD * DMODEL, wT, HD, DMODEL);
        launch_gemm(stream, oh, HD, ZB, ZB, wT, HD, ZB, ZB,
                    xf, nullptr, DMODEL, ZB, ZB, cbo + (size_t)i * DMODEL, xf,
                    TKNS, DMODEL, HD, 1, 1, 1.0f, 0);
    }

    hipMemcpyAsync(d_out, xf, (size_t)TKNS * DMODEL * sizeof(float),
                   hipMemcpyDeviceToDevice, stream);
}
